// MACE_51573967290533
// MI455X (gfx1250) — matrix operand order repaired and checked by reading, not compiled, not run on hardware
//
#include <hip/hip_runtime.h>
#include <hip/hip_bf16.h>
#include <math.h>

typedef __attribute__((ext_vector_type(16))) __bf16 v16bf;
typedef __attribute__((ext_vector_type(8)))  float  v8f;

#define Nn 10000
#define Ee 160000
#define Gg 20
#define Zz 10

// fast silu: x * rcp(1+exp(-x)) -- v_exp_f32 + v_rcp_f32, no IEEE div chain
__device__ __forceinline__ float silu_f(float x) {
    return x * __builtin_amdgcn_rcpf(1.0f + __expf(-x));
}

// ---------------------------------------------------------------------------
// WMMA fragment loaders (bf16, 16x16x32 shape)
// A: [16][64] LDS activations, row = edge, col = feature. Lane l holds row
//    l&15; K = (l>>4)*16 + i.
// B: transposed LDS weights wt[n][k] (row stride = wstride). Lane l holds
//    column (l&15)+n0; K = (l>>4)*16 + i.
// ---------------------------------------------------------------------------
__device__ __forceinline__ v16bf loadA(const __bf16* hin, int lane, int k0)
{
    const int rowA = lane & 15;
    const int kbA  = (lane >> 4) * 16;
    v16bf a;
#pragma unroll
    for (int i = 0; i < 16; ++i) {
        const int Ka = k0 + kbA + i;
        a[i] = hin[rowA * 64 + Ka];
    }
    return a;
}

__device__ __forceinline__ v16bf loadB(const __bf16* wt, int wstride, int n0,
                                       int k0, int lane)
{
    const int nB  = (lane & 15) + n0;
    const int kbB = (lane >> 4) * 16;
    v16bf b;
#pragma unroll
    for (int i = 0; i < 16; ++i) b[i] = wt[nB * wstride + k0 + kbB + i];
    return b;
}

// ---------------------------------------------------------------------------
// Edge pipeline: geometry -> 4-layer radial MLP (WMMA bf16) -> scatter-add
// block = 64 threads (2 waves); each wave handles 8 groups of 16 edges.
// ---------------------------------------------------------------------------
__launch_bounds__(64)
__global__ void mace_edge_kernel(const float* __restrict__ pos,
                                 const int*   __restrict__ eidx,
                                 const float* __restrict__ feats,
                                 float*       __restrict__ Abuf,
                                 const float* __restrict__ W0,
                                 const float* __restrict__ W1,
                                 const float* __restrict__ W2,
                                 const float* __restrict__ W3)
{
    __shared__ __bf16 Wt0[64][32];     // 4 KB  (8->64, zero padded K)
    __shared__ __bf16 Wt1[64][64];     // 8 KB
    __shared__ __bf16 Wt2[64][64];     // 8 KB
    __shared__ __bf16 Wt3[256][64];    // 32 KB
    __shared__ float  ytile[2][16][16];// 2 KB
    __shared__ __bf16 hstA[2][16][64]; // 4 KB
    __shared__ __bf16 hstB[2][16][64]; // 4 KB
    __shared__ int    sidx[2][16];
    __shared__ int    ridx[2][16];

    const int tid  = threadIdx.x;
    const int w    = tid >> 5;
    const int lane = tid & 31;

    // ---- stage weights: global f32 [k][n] -> LDS bf16 transposed [n][k] ----
    for (int idx = tid; idx < 64 * 32; idx += 64) {
        const int n = idx >> 5, kk = idx & 31;
        Wt0[n][kk] = (__bf16)((kk < 8) ? W0[kk * 64 + n] : 0.0f);
    }
    for (int idx = tid; idx < 64 * 64; idx += 64) {
        const int n = idx >> 6, k = idx & 63;
        Wt1[n][k] = (__bf16)W1[k * 64 + n];
        Wt2[n][k] = (__bf16)W2[k * 64 + n];
    }
    for (int idx = tid; idx < 256 * 64; idx += 64) {
        const int n = idx >> 6, k = idx & 63;
        Wt3[n][k] = (__bf16)W3[k * 256 + n];
    }
    __syncthreads();

    const int rbase = (lane >> 4) * 8;   // D: rows held by this lane
    const int ncol  = lane & 15;         // D: column within tile

    for (int g = 0; g < 8; ++g) {
        __syncthreads();  // protect LDS tiles from previous iteration readers
        const int ebase = blockIdx.x * 256 + w * 128 + g * 16;

        // ---------------- geometry: Y[16] and radial basis ----------------
        if (lane < 16) {
            const int e  = ebase + lane;
            const int s  = eidx[e];
            const int rv = eidx[Ee + e];
            sidx[w][lane] = s;
            ridx[w][lane] = rv;
            float x = pos[rv * 3 + 0] - pos[s * 3 + 0];
            float y = pos[rv * 3 + 1] - pos[s * 3 + 1];
            float z = pos[rv * 3 + 2] - pos[s * 3 + 2];
            const float r   = sqrtf(x * x + y * y + z * z + 1e-12f);
            const float inv = __builtin_amdgcn_rcpf(r);
            x *= inv; y *= inv; z *= inv;
            const float s3 = 1.7320508f, s5 = 2.2360680f, s15 = 3.8729833f;
            const float s7 = 2.6457513f;
            const float c31 = 1.6201852f, c32 = 10.246951f, c33 = 2.0916500f;
            float* yt = ytile[w][lane];
            yt[0]  = 1.0f;
            yt[1]  = s3 * x;   yt[2] = s3 * y;   yt[3] = s3 * z;
            yt[4]  = s15 * x * y;
            yt[5]  = s15 * y * z;
            yt[6]  = 0.5f * s5 * (3.0f * z * z - 1.0f);
            yt[7]  = s15 * x * z;
            yt[8]  = 0.5f * s15 * (x * x - y * y);
            yt[9]  = c33 * y * (3.0f * x * x - y * y);
            yt[10] = c32 * x * y * z;
            yt[11] = c31 * y * (5.0f * z * z - 1.0f);
            yt[12] = 0.5f * s7 * z * (5.0f * z * z - 3.0f);
            yt[13] = c31 * x * (5.0f * z * z - 1.0f);
            yt[14] = 0.5f * c32 * z * (x * x - y * y);
            yt[15] = c33 * x * (x * x - 3.0f * y * y);
            // Bessel radial basis with polynomial envelope
            const float u = r * 0.2f;
            float env = 0.0f;
            if (u < 1.0f) {
                const float u5 = u * u * u * u * u;
                env = 1.0f - 21.0f * u5 + 35.0f * u5 * u - 15.0f * u5 * u * u;
            }
            const float coef = 0.63245553f * inv * env;  // sqrt(2/R_MAX)/r*env
#pragma unroll
            for (int n = 0; n < 8; ++n)
                hstA[w][lane][n] = (__bf16)(coef * __sinf((float)(n + 1) * 3.14159265f * u));
#pragma unroll
            for (int n = 8; n < 32; ++n) hstA[w][lane][n] = (__bf16)0.0f;
        }
        __syncthreads();

        // ---------------- layer 0: [16x8(pad32)] @ [32x64] ----------------
        {
            const v16bf a0 = loadA(&hstA[w][0][0], lane, 0);
#pragma unroll
            for (int tn = 0; tn < 4; ++tn) {
                v8f c = {};
                const v16bf b0 = loadB(&Wt0[0][0], 32, tn * 16, 0, lane);
                c = __builtin_amdgcn_wmma_f32_16x16x32_bf16(false, a0, false, b0,
                                                            (short)0, c, false, false);
#pragma unroll
                for (int r = 0; r < 8; ++r)
                    hstB[w][rbase + r][tn * 16 + ncol] = (__bf16)silu_f(c[r]);
            }
        }
        __syncthreads();

        // ---------------- layer 1: [16x64] @ [64x64] ----------------
        {
            const v16bf a0 = loadA(&hstB[w][0][0], lane, 0);
            const v16bf a1 = loadA(&hstB[w][0][0], lane, 32);
#pragma unroll
            for (int tn = 0; tn < 4; ++tn) {
                v8f c = {};
                const v16bf b0 = loadB(&Wt1[0][0], 64, tn * 16, 0, lane);
                c = __builtin_amdgcn_wmma_f32_16x16x32_bf16(false, a0, false, b0,
                                                            (short)0, c, false, false);
                const v16bf b1 = loadB(&Wt1[0][0], 64, tn * 16, 32, lane);
                c = __builtin_amdgcn_wmma_f32_16x16x32_bf16(false, a1, false, b1,
                                                            (short)0, c, false, false);
#pragma unroll
                for (int r = 0; r < 8; ++r)
                    hstA[w][rbase + r][tn * 16 + ncol] = (__bf16)silu_f(c[r]);
            }
        }
        __syncthreads();

        // ---------------- layer 2: [16x64] @ [64x64] ----------------
        {
            const v16bf a0 = loadA(&hstA[w][0][0], lane, 0);
            const v16bf a1 = loadA(&hstA[w][0][0], lane, 32);
#pragma unroll
            for (int tn = 0; tn < 4; ++tn) {
                v8f c = {};
                const v16bf b0 = loadB(&Wt2[0][0], 64, tn * 16, 0, lane);
                c = __builtin_amdgcn_wmma_f32_16x16x32_bf16(false, a0, false, b0,
                                                            (short)0, c, false, false);
                const v16bf b1 = loadB(&Wt2[0][0], 64, tn * 16, 32, lane);
                c = __builtin_amdgcn_wmma_f32_16x16x32_bf16(false, a1, false, b1,
                                                            (short)0, c, false, false);
#pragma unroll
                for (int r = 0; r < 8; ++r)
                    hstB[w][rbase + r][tn * 16 + ncol] = (__bf16)silu_f(c[r]);
            }
        }
        __syncthreads();

        // ------- layer 3: [16x64] @ [64x256] fused with message scatter -------
        {
            const v16bf a0 = loadA(&hstB[w][0][0], lane, 0);
            const v16bf a1 = loadA(&hstB[w][0][0], lane, 32);
#pragma unroll
            for (int tn = 0; tn < 16; ++tn) {
                v8f c = {};
                const v16bf b0 = loadB(&Wt3[0][0], 64, tn * 16, 0, lane);
                c = __builtin_amdgcn_wmma_f32_16x16x32_bf16(false, a0, false, b0,
                                                            (short)0, c, false, false);
                const v16bf b1 = loadB(&Wt3[0][0], 64, tn * 16, 32, lane);
                c = __builtin_amdgcn_wmma_f32_16x16x32_bf16(false, a1, false, b1,
                                                            (short)0, c, false, false);
                const int cc = tn * 16 + ncol;  // output column = k*4 + l
                const int k  = cc >> 2;
                const int l  = cc & 3;
                const int m0 = l * l, m1 = (l + 1) * (l + 1);
#pragma unroll
                for (int r = 0; r < 8; ++r) {
                    const int row = rbase + r;
                    const int s   = sidx[w][row];
                    const int rv  = ridx[w][row];
                    const float f = feats[s * 64 + k] * 0.0625f * c[r]; // /AVG_NEIGH
                    float* dst = Abuf + (size_t)rv * 1024 + k * 16;
                    for (int m = m0; m < m1; ++m)
                        __hip_atomic_fetch_add(dst + m, f * ytile[w][row][m],
                                               __ATOMIC_RELAXED,
                                               __HIP_MEMORY_SCOPE_AGENT);
                }
            }
        }
    }
}

// ---------------------------------------------------------------------------
// Node update: per-m mix, invariants, species gate, W_out, readout.
// One 64-thread block per node.
// ---------------------------------------------------------------------------
__launch_bounds__(64)
__global__ void mace_node_kernel(const float* __restrict__ Abuf,
                                 float* __restrict__ feats,
                                 float* __restrict__ node_en,
                                 const float* __restrict__ attrs_g,
                                 const float* __restrict__ Wmix,
                                 const float* __restrict__ Wz,
                                 const float* __restrict__ Wout,
                                 const float* __restrict__ w_read1,
                                 const float* __restrict__ W_r1,
                                 const float* __restrict__ w_r2,
                                 int t)
{
    __shared__ float sA[64][16];
    __shared__ float sattr[Zz];
    __shared__ float bsh[64];
    __shared__ float fsh[64];
    __shared__ float red[64];

    const int n = blockIdx.x;
    const int c = threadIdx.x;

    for (int idx = c; idx < 1024; idx += 64)
        sA[idx >> 4][idx & 15] = Abuf[(size_t)n * 1024 + idx];
    if (c < Zz) sattr[c] = attrs_g[n * Zz + c];
    __syncthreads();

    const int lofm[16] = {0,1,1,1,2,2,2,2,2,3,3,3,3,3,3,3};
    float amix[16];
    float inv2 = 0.0f;
#pragma unroll
    for (int m = 0; m < 16; ++m) {
        const float* Wm = Wmix + lofm[m] * 4096;  // [k][c]
        float acc = 0.0f;
        for (int k = 0; k < 64; ++k) acc += sA[k][m] * Wm[k * 64 + c];
        amix[m] = acc;
        inv2 += acc * acc;
    }
    const float s1 = amix[0];

    float za = 0.0f, zb = 0.0f, zc = 0.0f;
#pragma unroll
    for (int z = 0; z < Zz; ++z) {
        const float a = sattr[z];
        za += a * Wz[z * 64 + c];
        zb += a * Wz[640 + z * 64 + c];
        zc += a * Wz[1280 + z * 64 + c];
    }
    const float b = za * s1 + zb * inv2 + zc * (s1 * inv2);
    bsh[c] = b;
    __syncthreads();

    float f = 0.0f;
    for (int j = 0; j < 64; ++j) f += bsh[j] * Wout[j * 64 + c];
    feats[n * 64 + c] = f;
    fsh[c] = f;
    red[c] = 0.0f;
    __syncthreads();

    if (t == 0) {
        red[c] = f * w_read1[c];
    } else if (c < 16) {
        float p = 0.0f;
        for (int j = 0; j < 64; ++j) p += fsh[j] * W_r1[j * 16 + c];
        red[c] = silu_f(p) * w_r2[c];
    }
    __syncthreads();
    if (c == 0) {
        float sum = 0.0f;
        for (int j = 0; j < 64; ++j) sum += red[j];
        node_en[n] += sum;
    }
}

// ---------------------------------------------------------------------------
__global__ void mace_init_kernel(const float* __restrict__ attrs,
                                 const float* __restrict__ W_embed,
                                 const float* __restrict__ ae,
                                 float* __restrict__ feats,
                                 float* __restrict__ node_en)
{
    const int tid = blockIdx.x * blockDim.x + threadIdx.x;
    if (tid >= Nn * 64) return;
    const int n = tid >> 6, k = tid & 63;
    float f = 0.0f;
    for (int z = 0; z < Zz; ++z) f += attrs[n * Zz + z] * W_embed[z * 64 + k];
    feats[tid] = f;
    if (k == 0) {
        float e = 0.0f;
        for (int z = 0; z < Zz; ++z) e += attrs[n * Zz + z] * ae[z];
        node_en[n] = e;
    }
}

__global__ void mace_reduce_kernel(const float* __restrict__ node_en,
                                   const int* __restrict__ batch,
                                   float* __restrict__ out)
{
    const int n = blockIdx.x * blockDim.x + threadIdx.x;
    if (n >= Nn) return;
    __hip_atomic_fetch_add(&out[batch[n]], node_en[n],
                           __ATOMIC_RELAXED, __HIP_MEMORY_SCOPE_AGENT);
}

// ---------------------------------------------------------------------------
extern "C" void kernel_launch(void* const* d_in, const int* in_sizes, int n_in,
                              void* d_out, int out_size, void* d_ws, size_t ws_size,
                              hipStream_t stream)
{
    const float* attrs   = (const float*)d_in[0];
    const float* pos     = (const float*)d_in[1];
    const int*   eidx    = (const int*)  d_in[2];
    const int*   batch   = (const int*)  d_in[3];
    const float* ae      = (const float*)d_in[4];
    const float* W_embed = (const float*)d_in[5];
    const float* rW0     = (const float*)d_in[6];
    const float* rW1     = (const float*)d_in[7];
    const float* rW2     = (const float*)d_in[8];
    const float* rW3     = (const float*)d_in[9];
    const float* Wmix    = (const float*)d_in[10];
    const float* Wz      = (const float*)d_in[11];
    const float* Wout    = (const float*)d_in[12];
    const float* w_read1 = (const float*)d_in[13];
    const float* W_r1    = (const float*)d_in[14];
    const float* w_r2    = (const float*)d_in[15];

    float* Abuf    = (float*)d_ws;                     // N*64*16 = 40.96 MB
    float* feats   = Abuf  + (size_t)Nn * 1024;        // N*64
    float* node_en = feats + (size_t)Nn * 64;          // N
    float* out     = (float*)d_out;

    mace_init_kernel<<<(Nn * 64 + 255) / 256, 256, 0, stream>>>(attrs, W_embed, ae,
                                                                feats, node_en);
    for (int t = 0; t < 2; ++t) {
        hipMemsetAsync(Abuf, 0, (size_t)Nn * 1024 * sizeof(float), stream);
        mace_edge_kernel<<<Ee / 256, 64, 0, stream>>>(
            pos, eidx, feats, Abuf,
            rW0 + t * 8 * 64, rW1 + t * 4096, rW2 + t * 4096, rW3 + t * 64 * 256);
        mace_node_kernel<<<Nn, 64, 0, stream>>>(
            Abuf, feats, node_en, attrs,
            Wmix + t * 16384, Wz + t * 1920, Wout + t * 4096,
            w_read1, W_r1, w_r2, t);
    }
    hipMemsetAsync(out, 0, Gg * sizeof(float), stream);
    mace_reduce_kernel<<<(Nn + 255) / 256, 256, 0, stream>>>(node_en, batch, out);
}
